// Att_dire_88545045774672
// MI455X (gfx1250) — matrix operand order repaired and checked by reading, not compiled, not run on hardware
//
#include <hip/hip_runtime.h>

// ---------------------------------------------------------------------------
// GAT-style attention aggregation, reduced to one f16 WMMA GEMM on gfx1250.
//   E = relu(h @ P)                       [4096, 3]
//   w[j,h] = exp(E[j,h] - max_j E[.,h])   (global per-head shift: softmax-safe)
//   Y[j, h*512+d] = w[j,h]*h[j,d]; extra cols: w[:,0..2], ones; pad -> 1664
//   Z = G @ Y   (f16 WMMA, f32 accumulate)  [4096, 1664]
//   out[i,d] = Z[i,1539] * sum_h Z[i,h*512+d] / Z[i,1536+h]
// ---------------------------------------------------------------------------

typedef _Float16 v16h __attribute__((ext_vector_type(16)));
typedef _Float16 v8h  __attribute__((ext_vector_type(8)));
typedef float    v8f  __attribute__((ext_vector_type(8)));

#define NN    4096
#define DD    512
#define HH    3
#define YW    1664          // 1536 data cols + 3 (S) + 1 (rowsum) + zero pad
#define KT    32            // K per WMMA step
#define NKT   (NN / KT)     // 128 K-steps
#define LDSW  40            // LDS row stride in halves (80 B): conflict-free b128

// -------- CDNA5 async global->LDS copy (ISA §15.18.3 opcode 98), inline asm --
// VDST = VGPR holding LDS byte address; VADDR = 64-bit global address; GV mode.
static __device__ __forceinline__ void async_copy_b128(const _Float16* gsrc,
                                                       _Float16* ldst) {
  const unsigned loff = (unsigned)(uintptr_t)(void*)ldst;    // low32 of flat == AS3 offset
  const unsigned long long ga = (unsigned long long)(uintptr_t)(const void*)gsrc;
  asm volatile("global_load_async_to_lds_b128 %0, %1, off"
               :: "v"(loff), "v"(ga) : "memory");
}

static __device__ __forceinline__ void wait_async0() {
#if defined(__has_builtin)
#if __has_builtin(__builtin_amdgcn_s_wait_asynccnt)
  __builtin_amdgcn_s_wait_asynccnt(0);
#else
  asm volatile("s_wait_asynccnt 0x0" ::: "memory");
#endif
#else
  asm volatile("s_wait_asynccnt 0x0" ::: "memory");
#endif
}

// ---------------------------------------------------------------- init mh ---
__global__ __launch_bounds__(32) void init_mh(int* mhbits) {
  if (threadIdx.x < 4) mhbits[threadIdx.x] = 0;   // 0.0f bits; E >= 0 after relu
}

// ------------------------------------------------- E = relu(h@P), head max ---
__global__ __launch_bounds__(256) void compute_e(const float* __restrict__ hmat,
                                                 const float* __restrict__ P,
                                                 float* __restrict__ E,
                                                 int* __restrict__ mhbits) {
  __shared__ float Ps[DD * HH];
  const int t = threadIdx.x;
  for (int i = t; i < DD * HH; i += 256) Ps[i] = P[i];
  __syncthreads();

  const int j = blockIdx.x * 256 + t;
  const float* hr = hmat + (size_t)j * DD;
  float a0 = 0.f, a1 = 0.f, a2 = 0.f;
#pragma unroll 8
  for (int d = 0; d < DD; ++d) {
    const float hv = hr[d];
    a0 += hv * Ps[d * 3 + 0];
    a1 += hv * Ps[d * 3 + 1];
    a2 += hv * Ps[d * 3 + 2];
  }
  a0 = fmaxf(a0, 0.f); a1 = fmaxf(a1, 0.f); a2 = fmaxf(a2, 0.f);
  E[j * 3 + 0] = a0; E[j * 3 + 1] = a1; E[j * 3 + 2] = a2;
  // floats >= 0: int-bit atomicMax == float max
  atomicMax(&mhbits[0], __float_as_int(a0));
  atomicMax(&mhbits[1], __float_as_int(a1));
  atomicMax(&mhbits[2], __float_as_int(a2));
}

// --------------------------------------- G (binary f32) -> f16, exact -------
__global__ __launch_bounds__(256) void convert_g(const float* __restrict__ G,
                                                 _Float16* __restrict__ Gh) {
  const size_t i = ((size_t)blockIdx.x * 256 + threadIdx.x) * 8;
  const float4 a = *(const float4*)(G + i);
  const float4 b = *(const float4*)(G + i + 4);
  v8h o;
  o[0] = (_Float16)a.x; o[1] = (_Float16)a.y; o[2] = (_Float16)a.z; o[3] = (_Float16)a.w;
  o[4] = (_Float16)b.x; o[5] = (_Float16)b.y; o[6] = (_Float16)b.z; o[7] = (_Float16)b.w;
  *(v8h*)(Gh + i) = o;
}

// ---------------------------------- Yt[n][j] = w[j,h]*h[j,d]  (transposed) ---
__global__ __launch_bounds__(256) void build_yt(const float* __restrict__ hmat,
                                                const float* __restrict__ E,
                                                const int* __restrict__ mhbits,
                                                _Float16* __restrict__ Yt) {
  __shared__ float tile[32][33];
  __shared__ float wsh[3][32];
  const int tx = threadIdx.x, ty = threadIdx.y;          // block (32, 8)
  const int d0 = blockIdx.x * 32, j0 = blockIdx.y * 32;

#pragma unroll
  for (int k = 0; k < 4; ++k)
    tile[ty + 8 * k][tx] = hmat[(size_t)(j0 + ty + 8 * k) * DD + d0 + tx];

  const int t = ty * 32 + tx;
  if (t < 96) {
    const int jj = t & 31, hh = t >> 5;
    wsh[hh][jj] = expf(E[(j0 + jj) * 3 + hh] - __int_as_float(mhbits[hh]));
  }
  __syncthreads();

#pragma unroll
  for (int hh = 0; hh < 3; ++hh) {
    const float w = wsh[hh][tx];                          // tx indexes j here
#pragma unroll
    for (int k = 0; k < 4; ++k) {
      const int d = d0 + ty + 8 * k;
      Yt[(size_t)(hh * DD + d) * NN + j0 + tx] = (_Float16)(w * tile[tx][ty + 8 * k]);
    }
  }
}

// ----------------------- extra columns: w (3), ones (1), zero pad (124) ------
__global__ __launch_bounds__(256) void pad_yt(const float* __restrict__ E,
                                              const int* __restrict__ mhbits,
                                              _Float16* __restrict__ Yt) {
  const int j = blockIdx.x * 256 + threadIdx.x;
  const int c = blockIdx.y;                               // 0..127
  float v;
  if (c < 3)       v = expf(E[j * 3 + c] - __int_as_float(mhbits[c]));
  else if (c == 3) v = 1.0f;
  else             v = 0.0f;
  Yt[(size_t)(1536 + c) * NN + j] = (_Float16)v;
}

// ----- shared WMMA micro-kernel: 8 wmma on buf `cur`, wave tile 64x32 --------
static __device__ __forceinline__ void wmma_step(const _Float16* Abc,
                                                 const _Float16* Bbc,
                                                 int wm, int wn, int h2, int l16,
                                                 v8f acc[4][2]) {
  v16h bf0, bf1;
  {
    const int nrow0 = wn * 32 + l16;
    bf0 = *(const v16h*)(&Bbc[nrow0 * LDSW + h2 * 16]);
    bf1 = *(const v16h*)(&Bbc[(nrow0 + 16) * LDSW + h2 * 16]);
  }
#pragma unroll
  for (int mm = 0; mm < 4; ++mm) {
    const int arow = wm * 64 + mm * 16 + l16;
    const v8h alo = *(const v8h*)(&Abc[arow * LDSW + h2 * 16]);
    const v8h ahi = *(const v8h*)(&Abc[arow * LDSW + h2 * 16 + 8]);
    v16h af;
#pragma unroll
    for (int e = 0; e < 8; ++e) { af[e] = alo[e]; af[8 + e] = ahi[e]; }
    acc[mm][0] = __builtin_amdgcn_wmma_f32_16x16x32_f16(
        false, af, false, bf0, (short)0, acc[mm][0], false, false);
    acc[mm][1] = __builtin_amdgcn_wmma_f32_16x16x32_f16(
        false, af, false, bf1, (short)0, acc[mm][1], false, false);
  }
}

static __device__ __forceinline__ void store_c(float* Z, int m0, int n0,
                                               int wm, int wn, int h2, int l16,
                                               v8f acc[4][2]) {
#pragma unroll
  for (int mm = 0; mm < 4; ++mm) {
    const int row = m0 + wm * 64 + mm * 16 + h2 * 8;
#pragma unroll
    for (int nn = 0; nn < 2; ++nn) {
      const int col = n0 + wn * 32 + nn * 16 + l16;
      float* zp = Z + (size_t)row * YW + col;
#pragma unroll
      for (int v = 0; v < 8; ++v)
        zp[(size_t)v * YW] = acc[mm][nn][v];
    }
  }
}

// --------------- Z = Gh @ Y : both tiles via async global->LDS ---------------
__global__ __launch_bounds__(256) void gemm_pre(const _Float16* __restrict__ Gh,
                                                const _Float16* __restrict__ Yt,
                                                float* __restrict__ Z) {
  __shared__ __align__(16) _Float16 Ab[2][128 * LDSW];
  __shared__ __align__(16) _Float16 Bb[2][128 * LDSW];

  const int tid  = threadIdx.x;
  const int lane = tid & 31;
  const int wave = tid >> 5;
  const int h2   = (lane >> 4) & 1;
  const int l16  = lane & 15;
  const int wm   = wave >> 2;
  const int wn   = wave & 3;

  const int m0 = blockIdx.y * 128;
  const int n0 = blockIdx.x * 128;

  const int lr = tid >> 1;
  const int lc = (tid & 1) * 16;
  const _Float16* aptr = Gh + (size_t)(m0 + lr) * NN + lc;
  const _Float16* bptr = Yt + (size_t)(n0 + lr) * NN + lc;

  _Float16* const al0 = &Ab[0][lr * LDSW + lc];
  _Float16* const al1 = &Ab[1][lr * LDSW + lc];
  _Float16* const bl0 = &Bb[0][lr * LDSW + lc];
  _Float16* const bl1 = &Bb[1][lr * LDSW + lc];

  v8f acc[4][2];
#pragma unroll
  for (int mm = 0; mm < 4; ++mm)
#pragma unroll
    for (int nn = 0; nn < 2; ++nn)
      acc[mm][nn] = (v8f){0.f, 0.f, 0.f, 0.f, 0.f, 0.f, 0.f, 0.f};

  // prologue: tile 0 -> buf 0
  async_copy_b128(aptr + 0, al0);
  async_copy_b128(aptr + 8, al0 + 8);
  async_copy_b128(bptr + 0, bl0);
  async_copy_b128(bptr + 8, bl0 + 8);

#pragma unroll 1
  for (int kt = 0; kt < NKT; ++kt) {
    const int cur = kt & 1;
    wait_async0();               // own async writes to buf `cur` complete
    __syncthreads();             // publish buf `cur`; readers of `cur^1` done
    if (kt + 1 < NKT) {
      const int k1 = (kt + 1) * KT;
      _Float16* adst = cur ? al0 : al1;
      _Float16* bdst = cur ? bl0 : bl1;
      async_copy_b128(aptr + k1,     adst);
      async_copy_b128(aptr + k1 + 8, adst + 8);
      async_copy_b128(bptr + k1,     bdst);
      async_copy_b128(bptr + k1 + 8, bdst + 8);
    }
    wmma_step(Ab[cur], Bb[cur], wm, wn, h2, l16, acc);
  }
  store_c(Z, m0, n0, wm, wn, h2, l16, acc);
}

// --------- Z = G @ Y : A converted f32->f16 on the fly (small-ws path) -------
__global__ __launch_bounds__(256, 1) void gemm_fly(const float* __restrict__ G,
                                                   const _Float16* __restrict__ Yt,
                                                   float* __restrict__ Z) {
  __shared__ __align__(16) _Float16 Ab[2][128 * LDSW];
  __shared__ __align__(16) _Float16 Bb[2][128 * LDSW];

  const int tid  = threadIdx.x;
  const int lane = tid & 31;
  const int wave = tid >> 5;
  const int h2   = (lane >> 4) & 1;
  const int l16  = lane & 15;
  const int wm   = wave >> 2;
  const int wn   = wave & 3;

  const int m0 = blockIdx.y * 128;
  const int n0 = blockIdx.x * 128;

  const int lr = tid >> 1;
  const int lc = (tid & 1) * 16;
  const float*    gptr = G  + (size_t)(m0 + lr) * NN + lc;
  const _Float16* bptr = Yt + (size_t)(n0 + lr) * NN + lc;

  _Float16* const bl0 = &Bb[0][lr * LDSW + lc];
  _Float16* const bl1 = &Bb[1][lr * LDSW + lc];
  _Float16* const al0 = &Ab[0][lr * LDSW + lc];
  _Float16* const al1 = &Ab[1][lr * LDSW + lc];

  v8f acc[4][2];
#pragma unroll
  for (int mm = 0; mm < 4; ++mm)
#pragma unroll
    for (int nn = 0; nn < 2; ++nn)
      acc[mm][nn] = (v8f){0.f, 0.f, 0.f, 0.f, 0.f, 0.f, 0.f, 0.f};

  float4 ga0, ga1, ga2, ga3;

  async_copy_b128(bptr + 0, bl0);
  async_copy_b128(bptr + 8, bl0 + 8);
  ga0 = *(const float4*)(gptr + 0);
  ga1 = *(const float4*)(gptr + 4);
  ga2 = *(const float4*)(gptr + 8);
  ga3 = *(const float4*)(gptr + 12);
  {
    v8h lo, hi;
    lo[0] = (_Float16)ga0.x; lo[1] = (_Float16)ga0.y; lo[2] = (_Float16)ga0.z; lo[3] = (_Float16)ga0.w;
    lo[4] = (_Float16)ga1.x; lo[5] = (_Float16)ga1.y; lo[6] = (_Float16)ga1.z; lo[7] = (_Float16)ga1.w;
    hi[0] = (_Float16)ga2.x; hi[1] = (_Float16)ga2.y; hi[2] = (_Float16)ga2.z; hi[3] = (_Float16)ga2.w;
    hi[4] = (_Float16)ga3.x; hi[5] = (_Float16)ga3.y; hi[6] = (_Float16)ga3.z; hi[7] = (_Float16)ga3.w;
    *(v8h*)(al0)     = lo;
    *(v8h*)(al0 + 8) = hi;
  }

#pragma unroll 1
  for (int kt = 0; kt < NKT; ++kt) {
    const int cur = kt & 1;
    const bool more = (kt + 1 < NKT);

    wait_async0();
    __syncthreads();

    if (more) {
      const int k1 = (kt + 1) * KT;
      _Float16* bdst = cur ? bl0 : bl1;
      async_copy_b128(bptr + k1,     bdst);
      async_copy_b128(bptr + k1 + 8, bdst + 8);
      ga0 = *(const float4*)(gptr + k1);
      ga1 = *(const float4*)(gptr + k1 + 4);
      ga2 = *(const float4*)(gptr + k1 + 8);
      ga3 = *(const float4*)(gptr + k1 + 12);
    }

    wmma_step(Ab[cur], Bb[cur], wm, wn, h2, l16, acc);

    if (more) {
      _Float16* adst = cur ? al0 : al1;
      v8h lo, hi;
      lo[0] = (_Float16)ga0.x; lo[1] = (_Float16)ga0.y; lo[2] = (_Float16)ga0.z; lo[3] = (_Float16)ga0.w;
      lo[4] = (_Float16)ga1.x; lo[5] = (_Float16)ga1.y; lo[6] = (_Float16)ga1.z; lo[7] = (_Float16)ga1.w;
      hi[0] = (_Float16)ga2.x; hi[1] = (_Float16)ga2.y; hi[2] = (_Float16)ga2.z; hi[3] = (_Float16)ga2.w;
      hi[4] = (_Float16)ga3.x; hi[5] = (_Float16)ga3.y; hi[6] = (_Float16)ga3.z; hi[7] = (_Float16)ga3.w;
      *(v8h*)(adst)     = lo;
      *(v8h*)(adst + 8) = hi;
    }
  }
  store_c(Z, m0, n0, wm, wn, h2, l16, acc);
}

// ------------------------------------------------------------- epilogue -----
__global__ __launch_bounds__(256) void epilogue(const float* __restrict__ Z,
                                                float* __restrict__ out) {
  const int i = blockIdx.y;
  const int d = blockIdx.x * 256 + threadIdx.x;
  const float* zr = Z + (size_t)i * YW;
  const float s0 = zr[1536], s1 = zr[1537], s2 = zr[1538], rs = zr[1539];
  const float i0 = s0 > 0.f ? 1.f / s0 : 0.f;
  const float i1 = s1 > 0.f ? 1.f / s1 : 0.f;
  const float i2 = s2 > 0.f ? 1.f / s2 : 0.f;
  out[(size_t)i * DD + d] =
      rs * (zr[d] * i0 + zr[DD + d] * i1 + zr[2 * DD + d] * i2);
}

// ------------------------------------------------------------- launcher -----
extern "C" void kernel_launch(void* const* d_in, const int* in_sizes, int n_in,
                              void* d_out, int out_size, void* d_ws, size_t ws_size,
                              hipStream_t stream) {
  const float* G    = (const float*)d_in[0];   // [4096,4096] binary f32
  const float* hmat = (const float*)d_in[1];   // [4096,512]  f32
  const float* P    = (const float*)d_in[2];   // [512,3]     f32
  float* out = (float*)d_out;                  // [4096,512]  f32

  char* ws = (char*)d_ws;
  const size_t SZ_GH = 33554432;   // 4096*4096*2
  const size_t SZ_YT = 13631488;   // 1664*4096*2
  const size_t SZ_Z  = 27262976;   // 4096*1664*4
  const size_t SZ_E  = 49152;      // 4096*3*4

  if (ws_size >= SZ_GH + SZ_YT + SZ_Z + SZ_E + 16) {
    // pre-converted adjacency: fully async GEMM
    _Float16* Gh  = (_Float16*)(ws);
    _Float16* Yt  = (_Float16*)(ws + SZ_GH);
    float*    Z   = (float*)(ws + SZ_GH + SZ_YT);
    float*    E   = (float*)(ws + SZ_GH + SZ_YT + SZ_Z);
    int*      mhb = (int*)(ws + SZ_GH + SZ_YT + SZ_Z + SZ_E);

    init_mh  <<<1, 32, 0, stream>>>(mhb);
    compute_e<<<16, 256, 0, stream>>>(hmat, P, E, mhb);
    convert_g<<<8192, 256, 0, stream>>>(G, Gh);
    build_yt <<<dim3(16, 128), dim3(32, 8), 0, stream>>>(hmat, E, mhb, Yt);
    pad_yt   <<<dim3(16, 128), 256, 0, stream>>>(E, mhb, Yt);
    gemm_pre <<<dim3(13, 32), 256, 0, stream>>>(Gh, Yt, Z);
    epilogue <<<dim3(2, 4096), 256, 0, stream>>>(Z, out);
  } else {
    // small-workspace path: convert adjacency on the fly inside the GEMM
    _Float16* Yt  = (_Float16*)(ws);
    float*    Z   = (float*)(ws + SZ_YT);
    float*    E   = (float*)(ws + SZ_YT + SZ_Z);
    int*      mhb = (int*)(ws + SZ_YT + SZ_Z + SZ_E);

    init_mh  <<<1, 32, 0, stream>>>(mhb);
    compute_e<<<16, 256, 0, stream>>>(hmat, P, E, mhb);
    build_yt <<<dim3(16, 128), dim3(32, 8), 0, stream>>>(hmat, E, mhb, Yt);
    pad_yt   <<<dim3(16, 128), 256, 0, stream>>>(E, mhb, Yt);
    gemm_fly <<<dim3(13, 32), 256, 0, stream>>>(G, Yt, Z);
    epilogue <<<dim3(2, 4096), 256, 0, stream>>>(Z, out);
  }
}
